// LSResNet_23063974379609
// MI455X (gfx1250) — matrix operand order repaired and checked by reading, not compiled, not run on hardware
//
#include <hip/hip_runtime.h>
#include <hip/hip_bf16.h>
#include <cstddef>

// ---------------------------------------------------------------------------
// LSResNet geodesic-conv layer for MI455X (gfx1250, wave32, WMMA).
//
// One workgroup (256 threads = 8 waves) per sample s.
//   Phase 1: stage mu/sigma (row f=0: setup tiles them identically over F),
//            per-vertex scalars, feat as f16 WMMA A-fragments, rho_g (f16 LDS).
//   Phase 2: for r in 0..15:
//              threads 0..199: th_g = exp(-(th-mu)^2/st), act = rho_g*th_g*mask,
//              normalize over 80 bins, write f16 WMMA B-fragments to LDS.
//              waves 0..4: desc(16x80) = feat(16x224) x act(224x80) via
//              7x v_wmma_f32_16x16x32_f16 each; scatter desc row r into the
//              A-fragments of the cf GEMM (M-dim = rotation index!).
//   Phase 3: for f in 0..4: stage W[f] as f16 B-fragments; waves 0..4 compute
//            cf(16x80) = desc(16x96) x W(96x80) (3 WMMAs per 16-col tile),
//            max over the 16 rotation rows (vgpr max + shfl_xor 16),
//            + bias, ReLU, store out[s, kbin*F + f].
// ---------------------------------------------------------------------------

typedef __attribute__((ext_vector_type(16))) _Float16 v16h;
typedef __attribute__((ext_vector_type(8)))  float    v8f;

namespace {
constexpr int F  = 5;
constexpr int K  = 80;        // gaussian bins
constexpr int R  = 16;        // rotations
constexpr int V  = 200;       // vertices per patch
constexpr int NT = K / 16;    // 5 N-tiles of 16 bins
constexpr int VC = 7;         // ceil(V/32) K-chunks for desc GEMM (pad to 224)
constexpr int KC = 3;         // ceil(K/32) K-chunks for cf GEMM  (pad to 96)
constexpr int RGS = 82;       // rho_g row stride in halves (bank-friendly)
constexpr float F_TWO_PI = 6.28318530717958647692f;
constexpr float F_DTH    = F_TWO_PI / 16.0f;
constexpr float F_EPS    = 1e-5f;

// dynamic-LDS layout (bytes)
constexpr size_t OFF_MUR   = 0;                            // K floats
constexpr size_t OFF_ISR   = OFF_MUR  + K * 4;             // K floats
constexpr size_t OFF_MUT   = OFF_ISR  + K * 4;             // K floats
constexpr size_t OFF_IST   = OFF_MUT  + K * 4;             // K floats
constexpr size_t OFF_RHO   = OFF_IST  + K * 4;             // V floats
constexpr size_t OFF_THETA = OFF_RHO  + V * 4;             // V floats
constexpr size_t OFF_MASK  = OFF_THETA+ V * 4;             // V floats
constexpr size_t OFF_AFEAT = OFF_MASK + V * 4;             // VC*512 halves
constexpr size_t OFF_ACT   = OFF_AFEAT+ VC * 512 * 2;      // VC*NT*512 halves
constexpr size_t OFF_DESC  = OFF_ACT  + VC * NT * 512 * 2; // F*KC*512 halves
constexpr size_t OFF_WFR   = OFF_DESC + F * KC * 512 * 2;  // KC*NT*512 halves
constexpr size_t OFF_RHOG  = OFF_WFR  + KC * NT * 512 * 2; // V*RGS halves
constexpr size_t SMEM_BYTES= OFF_RHOG + V * RGS * 2;       // ~110 KB dynamic LDS
}

// ISA 7.12.2 fragment layouts (wave32, 16-bit):
// A 16x32: lanes 0-15 hold row M with K {0..7,16..23}; lanes 16-31 K {8..15,24..31}
__device__ __forceinline__ int a_idx(int M, int Kk) {
  int g = Kk >> 3; // group of 8
  return (((M + ((g & 1) << 4)) << 4) | ((Kk & 7) + ((g >> 1) << 3)));
}
// B 32x16: lanes 0-15 hold col N with K {0..7,16..23}; lanes 16-31 K {8..15,24..31}
__device__ __forceinline__ int b_idx(int Kk, int N) {
  return (((N + (((Kk >> 3) & 1) << 4)) << 4) | ((Kk & 7) + (((Kk >> 4) & 1) << 3)));
}

__global__ __launch_bounds__(256, 1)
void lsresnet_geoconv_kernel(const float* __restrict__ x,
                             const float* __restrict__ mu_rho,
                             const float* __restrict__ sg_rho,
                             const float* __restrict__ mu_th,
                             const float* __restrict__ sg_th,
                             const float* __restrict__ Wc,
                             const float* __restrict__ bc,
                             float* __restrict__ out) {
  extern __shared__ char smem[];
  float*    s_mur   = (float*)(smem + OFF_MUR);
  float*    s_isr   = (float*)(smem + OFF_ISR);
  float*    s_mut   = (float*)(smem + OFF_MUT);
  float*    s_ist   = (float*)(smem + OFF_IST);
  float*    s_rho   = (float*)(smem + OFF_RHO);
  float*    s_theta = (float*)(smem + OFF_THETA);
  float*    s_mask  = (float*)(smem + OFF_MASK);
  _Float16* a_feat  = (_Float16*)(smem + OFF_AFEAT);
  _Float16* actf    = (_Float16*)(smem + OFF_ACT);
  _Float16* descf   = (_Float16*)(smem + OFF_DESC);
  _Float16* wfr     = (_Float16*)(smem + OFF_WFR);
  _Float16* rho_g   = (_Float16*)(smem + OFF_RHOG);

  const int s    = blockIdx.x;
  const int tid  = threadIdx.x;
  const int lane = tid & 31;
  const int wave = tid >> 5;

  // Warm L2 with this sample's W usage (gfx1250 global_prefetch_b8).
  for (int i = tid * 16; i < F * K * K; i += 256 * 16)
    __builtin_prefetch(Wc + i, 0, 1);

  // Phase 1a: mu/sigma row f=0 (rows tiled identically), zero all fragments
  // (covers pad rows: f>=5, v>=200, kbin>=80 so stale LDS/NaNs never leak).
  for (int k = tid; k < K; k += 256) {
    float sr = sg_rho[k], st = sg_th[k];
    s_mur[k] = mu_rho[k]; s_isr[k] = 1.0f / (sr * sr + F_EPS);
    s_mut[k] = mu_th[k];  s_ist[k] = 1.0f / (st * st + F_EPS);
  }
  for (int i = tid; i < VC * 512;      i += 256) a_feat[i] = (_Float16)0.0f;
  for (int i = tid; i < VC * NT * 512; i += 256) actf[i]   = (_Float16)0.0f;
  for (int i = tid; i < F * KC * 512;  i += 256) descf[i]  = (_Float16)0.0f;
  for (int i = tid; i < KC * NT * 512; i += 256) wfr[i]    = (_Float16)0.0f;
  __syncthreads();

  // Phase 1b: per-vertex scalars + feat into A-fragments (M=f, Kdim=v).
  if (tid < V) {
    const float* xv = x + ((size_t)s * V + tid) * 8;
    s_rho[tid]   = xv[5];
    s_theta[tid] = xv[6];
    s_mask[tid]  = xv[7];
    const int c = tid >> 5, kk = tid & 31;
    #pragma unroll
    for (int f = 0; f < F; ++f)
      a_feat[(c << 9) + a_idx(f, kk)] = (_Float16)xv[f];
  }
  __syncthreads();

  // Phase 1c: rotation-independent rho Gaussians (f16, reused by all 16 rots).
  for (int i = tid; i < V * K; i += 256) {
    const int v = i / K, k = i - v * K;
    const float d = s_rho[v] - s_mur[k];
    rho_g[v * RGS + k] = (_Float16)__expf(-d * d * s_isr[k]);
  }

  // Phase 2: rotations
  for (int r = 0; r < R; ++r) {
    __syncthreads(); // previous desc GEMM done reading actf
    if (tid < V) {
      const int v = tid, c = v >> 5, kk = v & 31;
      float th = s_theta[v] + (float)r * F_DTH;
      if (th >= F_TWO_PI) th -= F_TWO_PI;       // theta in [0,2pi), r*dth < 2pi
      const float mk = s_mask[v];
      float sum = 0.0f;
      #pragma unroll 4
      for (int k = 0; k < K; ++k) {
        const float d = th - s_mut[k];
        const float g = __expf(-d * d * s_ist[k]) * (float)rho_g[v * RGS + k] * mk;
        actf[((c * NT + (k >> 4)) << 9) + b_idx(kk, k & 15)] = (_Float16)g;
        sum += g;
      }
      const float scl = 1.0f / (sum + F_EPS);
      #pragma unroll 4
      for (int k = 0; k < K; ++k) {
        const int o = ((c * NT + (k >> 4)) << 9) + b_idx(kk, k & 15);
        actf[o] = (_Float16)((float)actf[o] * scl);
      }
    }
    __syncthreads();

    // desc(16x80) = feat(16x224) x act(224x80); wave w owns 16-bin tile w.
    if (wave < NT) {
      const int nt = wave;
      v8f acc = {};
      #pragma unroll
      for (int c = 0; c < VC; ++c) {
        v16h a = *(const v16h*)(a_feat + (c << 9) + lane * 16);
        v16h b = *(const v16h*)(actf + ((c * NT + nt) << 9) + lane * 16);
        acc = __builtin_amdgcn_wmma_f32_16x16x32_f16(
            false, a, false, b, (short)0, acc, false, false);
      }
      // Scatter desc row r into cf-GEMM A-fragments (M=rotation, Kdim=bin).
      // C/D layout: lane<16 holds rows 0..7 (= f 0..4 + zero pad) of col lane.
      if (lane < 16) {
        const int kb = nt * 16 + lane;
        const int kc = kb >> 5, kk2 = kb & 31;
        #pragma unroll
        for (int f = 0; f < F; ++f)
          descf[((f * KC + kc) << 9) + a_idx(r, kk2)] = (_Float16)acc[f];
      }
    }
  }

  // Phase 3: cf(16x80) = desc(16x96) x W_f(96x80); max over rotations; ReLU.
  for (int f = 0; f < F; ++f) {
    __syncthreads();
    for (int i = tid; i < K * K; i += 256) { // stage W[f] B-fragments (kb<80)
      const int kb = i / K, j = i - kb * K;
      wfr[(((kb >> 5) * NT + (j >> 4)) << 9) + b_idx(kb & 31, j & 15)] =
          (_Float16)Wc[((size_t)(f * K + kb)) * K + j];
    }
    __syncthreads();
    if (wave < NT) {
      const int nt = wave;
      v8f acc = {};
      #pragma unroll
      for (int kc = 0; kc < KC; ++kc) {
        v16h a = *(const v16h*)(descf + ((f * KC + kc) << 9) + lane * 16);
        v16h b = *(const v16h*)(wfr + ((kc * NT + nt) << 9) + lane * 16);
        acc = __builtin_amdgcn_wmma_f32_16x16x32_f16(
            false, a, false, b, (short)0, acc, false, false);
      }
      // rows = rotations: per-lane max over 8 vgprs, then fold lanes L<->L+16
      float m = acc[0];
      #pragma unroll
      for (int i = 1; i < 8; ++i) m = fmaxf(m, acc[i]);
      m = fmaxf(m, __shfl_xor(m, 16, 32));
      if (lane < 16) {
        const int kb = nt * 16 + lane;
        const float val = m + bc[f * K + kb];
        out[(size_t)s * (K * F) + kb * F + f] = fmaxf(val, 0.0f);
      }
    }
  }
}

extern "C" void kernel_launch(void* const* d_in, const int* in_sizes, int n_in,
                              void* d_out, int out_size, void* d_ws, size_t ws_size,
                              hipStream_t stream) {
  (void)n_in; (void)d_ws; (void)ws_size; (void)out_size;
  const float* x      = (const float*)d_in[0];
  const float* mu_rho = (const float*)d_in[1];
  const float* sg_rho = (const float*)d_in[2];
  const float* mu_th  = (const float*)d_in[3];
  const float* sg_th  = (const float*)d_in[4];
  const float* Wc     = (const float*)d_in[5];
  const float* bc     = (const float*)d_in[6];
  float* out = (float*)d_out;

  const int S = in_sizes[0] / (V * 8); // 1000 samples
  lsresnet_geoconv_kernel<<<dim3(S), dim3(256), SMEM_BYTES, stream>>>(
      x, mu_rho, sg_rho, mu_th, sg_th, Wc, bc, out);
}
